// B2_39015482917661
// MI455X (gfx1250) — matrix operand order repaired and checked by reading, not compiled, not run on hardware
//
#include <hip/hip_runtime.h>
#include <hip/hip_bf16.h>

// ---------------------------------------------------------------------------
// Problem constants (from reference)
// ---------------------------------------------------------------------------
#define BB      4
#define NN      512
#define INDIM   1024
#define HID     256
#define MID     512
#define GATE    1024      // 4*HID
#define OUTK    50
#define EMB     64
#define NPOS    30
#define WROW    1090      // 2*(MID+1) + EMB
#define ROWS    (BB*NN)   // 2048

typedef __attribute__((ext_vector_type(16))) __bf16 v16bf;
typedef __attribute__((ext_vector_type(8)))  __bf16 v8bf;
typedef __attribute__((ext_vector_type(8)))  float  v8f;
typedef __attribute__((ext_vector_type(4)))  float  v4f;

union V16U { v16bf v; v8bf h[2]; };

__device__ __forceinline__ v16bf ld16(const __bf16* p0, const __bf16* p1) {
  V16U u;
  u.h[0] = *(const v8bf*)p0;
  u.h[1] = *(const v8bf*)p1;
  return u.v;
}

__device__ __forceinline__ v8f wmma_bf16f32(v16bf a, v16bf b, v8f c) {
  // D(f32 16x16) = A(bf16 16x32) * B(bf16 32x16) + C
  return __builtin_amdgcn_wmma_f32_16x16x32_bf16(false, a, false, b,
                                                 (short)0, c, false, false);
}

__device__ __forceinline__ float sigf(float x) {
  return 1.0f / (1.0f + __expf(-x));
}

// ---------------------------------------------------------------------------
// Prep kernels
// ---------------------------------------------------------------------------

// Vectorized f32 -> bf16 conversion, 8 elements/thread (n must be mult of 8).
__global__ void k_cvt_bf16x8(const float* __restrict__ src,
                             __bf16* __restrict__ dst, int n8) {
  int i = blockIdx.x * blockDim.x + threadIdx.x;
  if (i >= n8) return;
  const v4f* s = (const v4f*)(src + (size_t)i * 8);
  v4f a = s[0], b = s[1];
  v8bf o;
  o[0] = (__bf16)a[0]; o[1] = (__bf16)a[1];
  o[2] = (__bf16)a[2]; o[3] = (__bf16)a[3];
  o[4] = (__bf16)b[0]; o[5] = (__bf16)b[1];
  o[6] = (__bf16)b[2]; o[7] = (__bf16)b[3];
  *(v8bf*)(dst + (size_t)i * 8) = o;
}

__global__ void k_addbias(const float* __restrict__ a,
                          const float* __restrict__ b,
                          float* __restrict__ o, int n) {
  int i = blockIdx.x * blockDim.x + threadIdx.x;
  if (i < n) o[i] = a[i] + b[i];
}

// Split W (50 x 1090) into bf16 Wh (50x512), Wt (50x512) and biases.
__global__ void k_extract_w(const float* __restrict__ W,
                            __bf16* __restrict__ Whb,
                            __bf16* __restrict__ Wtb,
                            float* __restrict__ bh,
                            float* __restrict__ bt) {
  int i = blockIdx.x * blockDim.x + threadIdx.x;
  if (i >= OUTK * MID) return;
  int k = i / MID, c = i % MID;
  Whb[k * MID + c] = (__bf16)W[k * WROW + c];
  Wtb[k * MID + c] = (__bf16)W[k * WROW + (MID + 1) + c];
  if (c == 0) {
    bh[k] = W[k * WROW + MID];
    bt[k] = W[k * WROW + 2 * MID + 1];
  }
}

// T[k][id] = sum_h emb[id][h] * W[k][1026+h]   (50 x 30 LUT)
__global__ void k_size_table(const float* __restrict__ emb,
                             const float* __restrict__ W,
                             float* __restrict__ T) {
  int i = blockIdx.x * blockDim.x + threadIdx.x;
  if (i >= OUTK * NPOS) return;
  int k = i / NPOS, id = i % NPOS;
  float s = 0.f;
  const float* e = emb + (size_t)id * EMB;
  const float* w = W + (size_t)k * WROW + 2 * (MID + 1);
  #pragma unroll 8
  for (int h = 0; h < EMB; ++h) s += e[h] * w[h];
  T[k * NPOS + id] = s;
}

// ---------------------------------------------------------------------------
// Generic WMMA GEMM:  C = act( A(MxK,bf16) * Bw(NxK,bf16)^T + bias[N] )
// One wave computes a (16*TM)x(16*TN) macro tile. M, K multiples of 32.
// OUT: 0 = f32 row-major [M][N]; 1 = bf16 row-major [M][N];
//      2 = f32 "tail-transposed" [(m/512)*OUTK + n][512] + (m%512)
// ---------------------------------------------------------------------------
enum { OUT_F32 = 0, OUT_BF16 = 1, OUT_TRANS = 2 };

template <int TM, int TN, bool LEAKY, int OUT>
__global__ __launch_bounds__(256) void k_gemm_bf16(
    const __bf16* __restrict__ A, const __bf16* __restrict__ Bw,
    const float* __restrict__ bias, float* __restrict__ Cf,
    __bf16* __restrict__ Cb, int M, int N, int K, int nTilesN) {
  int wid  = blockIdx.x * 8 + (threadIdx.x >> 5);
  int m32  = wid / nTilesN;          // macro-row  (TM*16 rows)
  int n32  = wid % nTilesN;          // macro-col  (TN*16 cols)
  int lane = threadIdx.x & 31;
  int half = lane >> 4;
  int l16  = lane & 15;

  v8f acc[TM][TN] = {};
  for (int k0 = 0; k0 < K; k0 += 32) {
    v16bf a[TM], b[TN];
    #pragma unroll
    for (int i = 0; i < TM; ++i) {
      const __bf16* ap = A + (size_t)(m32 * (TM * 16) + i * 16 + l16) * K
                           + k0 + half * 8;
      a[i] = ld16(ap, ap + 16);
    }
    #pragma unroll
    for (int j = 0; j < TN; ++j) {
      int col = n32 * (TN * 16) + j * 16 + l16;
      if (col >= N) col = N - 1;                 // clamped (store guarded)
      const __bf16* bp = Bw + (size_t)col * K + k0 + half * 8;
      b[j] = ld16(bp, bp + 16);
    }
    #pragma unroll
    for (int i = 0; i < TM; ++i)
      #pragma unroll
      for (int j = 0; j < TN; ++j)
        acc[i][j] = wmma_bf16f32(a[i], b[j], acc[i][j]);
  }

  #pragma unroll
  for (int i = 0; i < TM; ++i)
    #pragma unroll
    for (int j = 0; j < TN; ++j) {
      int n = n32 * (TN * 16) + j * 16 + l16;
      if (n >= N) continue;
      float bv = bias[n];
      #pragma unroll
      for (int r = 0; r < 8; ++r) {
        int m = m32 * (TM * 16) + i * 16 + r + half * 8;
        float v = acc[i][j][r] + bv;
        if (LEAKY) v = (v > 0.f) ? v : 0.01f * v;
        if (OUT == OUT_F32) {
          Cf[(size_t)m * N + n] = v;
        } else if (OUT == OUT_BF16) {
          Cb[(size_t)m * N + n] = (__bf16)v;
        } else {  // OUT_TRANS: store s_tail directly in [b][k][n] layout
          int bq = m >> 9, nn2 = m & (NN - 1);
          Cf[((size_t)(bq * OUTK + n)) * NN + nn2] = v;
        }
      }
    }
}

// ---------------------------------------------------------------------------
// LSTM recurrence. grid = {2} (dir), 512 threads (16 waves).
// Per step: g(16x1024) = hA(16x256,bf16) @ Whh^T via 512 WMMAs, then gate
// nonlinearity + state update. G holds precomputed x@Wih^T + biases.
//
// Residency plan (no global weight reads and no spills inside the loop):
//  - Whh K=0..127 staged once in LDS (256 KB of the 320 KB WGP LDS)
//  - Whh K=128..255 held in VGPRs (16 operands = 128 VGPRs per wave)
//  - cell state c lives in 2 registers per thread (static tid->(b,u) map)
//  - G rows for step t are prefetched during the WMMA phase
// ---------------------------------------------------------------------------
__global__ __launch_bounds__(512) void k_lstm(
    const __bf16* __restrict__ WhhBf,    // [2][GATE][HID]
    const float* __restrict__ G,         // [2][ROWS][GATE]
    __bf16* __restrict__ Xcat) {         // [ROWS][MID]
  int dir = blockIdx.x;
  const __bf16* Whh = WhhBf + (size_t)dir * GATE * HID;
  const float*  Gd  = G + (size_t)dir * ROWS * GATE;

  __shared__ __align__(16) __bf16 WhhLds[GATE * 128];  // K=0..127  (256 KB)
  __shared__ __align__(16) __bf16 hA[16][HID];         // WMMA-A rows (8 KB)
  __shared__ float gbuf[4][GATE];                      // gate staging (16 KB)

  int tid = threadIdx.x;

  // Stage Whh[:, 0:128] into LDS (16-byte chunks; 16 chunks per 128-elem row).
  for (int i = tid; i < GATE * 128 / 8; i += 512) {
    int c = i >> 4, chunk = i & 15;
    *(v8bf*)(WhhLds + c * 128 + chunk * 8) =
        *(const v8bf*)(Whh + (size_t)c * HID + chunk * 8);
  }
  for (int i = tid; i < 16 * HID; i += 512) ((__bf16*)hA)[i] = (__bf16)0.f;

  int wave = tid >> 5, lane = tid & 31;
  int half = lane >> 4, l16 = lane & 15;

  // Static update-phase ownership: this thread updates (b0,u) and (b0+2,u).
  int b0 = tid >> 8;            // 0 or 1
  int uu = tid & (HID - 1);
  float creg0 = 0.f, creg1 = 0.f;

  // Register-resident half: K=128..255 for this wave's 4 gate-column tiles.
  v16bf breg[4][4];
  #pragma unroll
  for (int j = 0; j < 4; ++j) {
    int col = wave * 64 + j * 16 + l16;
    #pragma unroll
    for (int kk = 0; kk < 4; ++kk) {
      const __bf16* bp = Whh + (size_t)col * HID + 128 + kk * 32 + half * 8;
      breg[j][kk] = ld16(bp, bp + 16);
    }
  }
  __syncthreads();

  for (int s = 0; s < NN; ++s) {
    int t = dir ? (NN - 1 - s) : s;

    // Prefetch this step's G gate rows so the update phase hits cache.
    const float* g0 = Gd + ((size_t)(b0 * NN + t)) * GATE + uu;
    const float* g1 = Gd + ((size_t)((b0 + 2) * NN + t)) * GATE + uu;
    #pragma unroll
    for (int q = 0; q < 4; ++q) {
      __builtin_prefetch(g0 + q * HID, 0, 3);
      __builtin_prefetch(g1 + q * HID, 0, 3);
    }

    v8f acc[4] = {};
    const __bf16* arow = &hA[l16][0];
    // K = 0..127 : B from LDS
    for (int kk = 0; kk < 4; ++kk) {
      int k0 = kk * 32;
      v16bf a = ld16(arow + k0 + half * 8, arow + k0 + 16 + half * 8);
      #pragma unroll
      for (int j = 0; j < 4; ++j) {
        int col = wave * 64 + j * 16 + l16;
        const __bf16* bp = WhhLds + (size_t)col * 128 + k0 + half * 8;
        v16bf b = ld16(bp, bp + 16);
        acc[j] = wmma_bf16f32(a, b, acc[j]);
      }
    }
    // K = 128..255 : B from registers
    #pragma unroll
    for (int kk = 0; kk < 4; ++kk) {
      int k0 = 128 + kk * 32;
      v16bf a = ld16(arow + k0 + half * 8, arow + k0 + 16 + half * 8);
      #pragma unroll
      for (int j = 0; j < 4; ++j)
        acc[j] = wmma_bf16f32(a, breg[j][kk], acc[j]);
    }

    // rows 0..3 of C are the live batches (lanes 0-15, VGPR r = row r)
    if (half == 0) {
      #pragma unroll
      for (int j = 0; j < 4; ++j) {
        int col = wave * 64 + j * 16 + l16;
        gbuf[0][col] = acc[j][0];
        gbuf[1][col] = acc[j][1];
        gbuf[2][col] = acc[j][2];
        gbuf[3][col] = acc[j][3];
      }
    }
    __syncthreads();

    // Gate nonlinearity + state update: thread owns (b0,uu) and (b0+2,uu).
    {
      float gi = gbuf[b0][uu]           + g0[0];
      float gf = gbuf[b0][HID + uu]     + g0[HID];
      float gg = gbuf[b0][2 * HID + uu] + g0[2 * HID];
      float go = gbuf[b0][3 * HID + uu] + g0[3 * HID];
      creg0 = sigf(gf) * creg0 + sigf(gi) * tanhf(gg);
      float h = sigf(go) * tanhf(creg0);
      hA[b0][uu] = (__bf16)h;
      Xcat[((size_t)(b0 * NN + t)) * MID + dir * HID + uu] = (__bf16)h;
    }
    {
      int b1 = b0 + 2;
      float gi = gbuf[b1][uu]           + g1[0];
      float gf = gbuf[b1][HID + uu]     + g1[HID];
      float gg = gbuf[b1][2 * HID + uu] + g1[2 * HID];
      float go = gbuf[b1][3 * HID + uu] + g1[3 * HID];
      creg1 = sigf(gf) * creg1 + sigf(gi) * tanhf(gg);
      float h = sigf(go) * tanhf(creg1);
      hA[b1][uu] = (__bf16)h;
      Xcat[((size_t)(b1 * NN + t)) * MID + dir * HID + uu] = (__bf16)h;
    }
    __syncthreads();
  }
}

// ---------------------------------------------------------------------------
// Final broadcast-add: out[b,k,m,n] = sHead[b,m,k] + stT[b,k,n] + T[k,id(m,n)]
// Pure HBM-write-bound (~210 MB). 128 threads x float4 covers one n-row.
// ---------------------------------------------------------------------------
__global__ __launch_bounds__(128) void k_fuse_out(
    const float* __restrict__ sHead,   // [ROWS][OUTK]
    const float* __restrict__ stT,     // [BB*OUTK][NN]
    const float* __restrict__ T,       // [OUTK][NPOS]
    float* __restrict__ out) {
  int bid = blockIdx.x;
  int m  = bid & (NN - 1);
  int kb = bid >> 9;
  int k  = kb % OUTK;
  int b  = kb / OUTK;

  __shared__ float Ts[NPOS];
  if (threadIdx.x < NPOS) Ts[threadIdx.x] = T[k * NPOS + threadIdx.x];
  __syncthreads();

  float shv = sHead[((size_t)(b * NN + m)) * OUTK + k];
  const float* tp = stT + ((size_t)(b * OUTK + k)) * NN;
  int n = threadIdx.x * 4;
  float4 tv = *(const float4*)(tp + n);

  int i0 = n + 0 - m + (NPOS / 2); i0 = i0 < 0 ? 0 : (i0 > NPOS - 1 ? NPOS - 1 : i0);
  int i1 = n + 1 - m + (NPOS / 2); i1 = i1 < 0 ? 0 : (i1 > NPOS - 1 ? NPOS - 1 : i1);
  int i2 = n + 2 - m + (NPOS / 2); i2 = i2 < 0 ? 0 : (i2 > NPOS - 1 ? NPOS - 1 : i2);
  int i3 = n + 3 - m + (NPOS / 2); i3 = i3 < 0 ? 0 : (i3 > NPOS - 1 ? NPOS - 1 : i3);

  float4 o;
  o.x = shv + tv.x + Ts[i0];
  o.y = shv + tv.y + Ts[i1];
  o.z = shv + tv.z + Ts[i2];
  o.w = shv + tv.w + Ts[i3];
  *(float4*)(out + ((((size_t)(b * OUTK + k)) * NN + m) * NN + n)) = o;
}

// ---------------------------------------------------------------------------
// Host-side orchestration
// ---------------------------------------------------------------------------
static inline size_t alignup(size_t x) { return (x + 255) & ~(size_t)255; }

extern "C" void kernel_launch(void* const* d_in, const int* in_sizes, int n_in,
                              void* d_out, int out_size, void* d_ws, size_t ws_size,
                              hipStream_t stream) {
  (void)in_sizes; (void)n_in; (void)out_size; (void)ws_size;
  const float* X      = (const float*)d_in[0];
  const float* Wih_f  = (const float*)d_in[1];
  const float* Whh_f  = (const float*)d_in[2];
  const float* bih_f  = (const float*)d_in[3];
  const float* bhh_f  = (const float*)d_in[4];
  const float* Wih_b  = (const float*)d_in[5];
  const float* Whh_b  = (const float*)d_in[6];
  const float* bih_b  = (const float*)d_in[7];
  const float* bhh_b  = (const float*)d_in[8];
  const float* W_head = (const float*)d_in[9];
  const float* b_head = (const float*)d_in[10];
  const float* W_tail = (const float*)d_in[11];
  const float* b_tail = (const float*)d_in[12];
  const float* embT   = (const float*)d_in[13];
  const float* W      = (const float*)d_in[14];
  float* out = (float*)d_out;

  char* ws = (char*)d_ws;
  size_t off = 0;
  auto take = [&](size_t bytes) { char* p = ws + off; off += alignup(bytes); return p; };

  __bf16* Xbf    = (__bf16*)take((size_t)ROWS * INDIM * 2);
  __bf16* WihFb  = (__bf16*)take((size_t)GATE * INDIM * 2);
  __bf16* WihBb  = (__bf16*)take((size_t)GATE * INDIM * 2);
  __bf16* WhhBf  = (__bf16*)take((size_t)2 * GATE * HID * 2);
  __bf16* WheadB = (__bf16*)take((size_t)MID * MID * 2);
  __bf16* WtailB = (__bf16*)take((size_t)MID * MID * 2);
  __bf16* Whb    = (__bf16*)take((size_t)OUTK * MID * 2);
  __bf16* Wtb    = (__bf16*)take((size_t)OUTK * MID * 2);
  float*  biasF  = (float*)take(GATE * 4);
  float*  biasB  = (float*)take(GATE * 4);
  float*  bh     = (float*)take(OUTK * 4);
  float*  bt     = (float*)take(OUTK * 4);
  float*  G      = (float*)take((size_t)2 * ROWS * GATE * 4);
  __bf16* Xcat   = (__bf16*)take((size_t)ROWS * MID * 2);
  __bf16* headB  = (__bf16*)take((size_t)ROWS * MID * 2);
  __bf16* tailB  = (__bf16*)take((size_t)ROWS * MID * 2);
  float*  sHead  = (float*)take((size_t)ROWS * OUTK * 4);
  float*  sTailT = (float*)take((size_t)BB * OUTK * NN * 4);
  float*  Ttab   = (float*)take((size_t)OUTK * NPOS * 4);

  auto cvt = [&](const float* s, __bf16* d, int n) {
    int n8 = n / 8;
    k_cvt_bf16x8<<<(n8 + 255) / 256, 256, 0, stream>>>(s, d, n8);
  };
  cvt(X,      Xbf,   ROWS * INDIM);
  cvt(Wih_f,  WihFb, GATE * INDIM);
  cvt(Wih_b,  WihBb, GATE * INDIM);
  cvt(Whh_f,  WhhBf,                      GATE * HID);
  cvt(Whh_b,  WhhBf + (size_t)GATE * HID, GATE * HID);
  cvt(W_head, WheadB, MID * MID);
  cvt(W_tail, WtailB, MID * MID);

  k_addbias<<<(GATE + 255) / 256, 256, 0, stream>>>(bih_f, bhh_f, biasF, GATE);
  k_addbias<<<(GATE + 255) / 256, 256, 0, stream>>>(bih_b, bhh_b, biasB, GATE);
  k_extract_w<<<(OUTK * MID + 255) / 256, 256, 0, stream>>>(W, Whb, Wtb, bh, bt);
  k_size_table<<<(OUTK * NPOS + 255) / 256, 256, 0, stream>>>(embT, W, Ttab);

  // G[dir] = X @ Wih^T + (bih+bhh)   M=2048 N=1024 K=1024, 32x64 macro tiles
  k_gemm_bf16<2, 4, false, OUT_F32><<<(64 * 16) / 8, 256, 0, stream>>>(
      Xbf, WihFb, biasF, G, nullptr, ROWS, GATE, INDIM, 16);
  k_gemm_bf16<2, 4, false, OUT_F32><<<(64 * 16) / 8, 256, 0, stream>>>(
      Xbf, WihBb, biasB, G + (size_t)ROWS * GATE, nullptr, ROWS, GATE, INDIM, 16);

  // Bidirectional LSTM recurrence (one block per direction)
  k_lstm<<<2, 512, 0, stream>>>(WhhBf, G, Xcat);

  // head/tail = leaky_relu(Xcat @ W^T + b)   M=2048 N=512 K=512, bf16 out
  k_gemm_bf16<2, 4, true, OUT_BF16><<<(64 * 8) / 8, 256, 0, stream>>>(
      Xcat, WheadB, b_head, nullptr, headB, ROWS, MID, MID, 8);
  k_gemm_bf16<2, 4, true, OUT_BF16><<<(64 * 8) / 8, 256, 0, stream>>>(
      Xcat, WtailB, b_tail, nullptr, tailB, ROWS, MID, MID, 8);

  // s_head / s_tail skinny GEMMs  M=2048 N=50 K=512 (the "+1" col is a bias)
  k_gemm_bf16<2, 2, false, OUT_F32><<<(64 * 2) / 8, 256, 0, stream>>>(
      headB, Whb, bh, sHead, nullptr, ROWS, OUTK, MID, 2);
  k_gemm_bf16<2, 2, false, OUT_TRANS><<<(64 * 2) / 8, 256, 0, stream>>>(
      tailB, Wtb, bt, sTailT, nullptr, ROWS, OUTK, MID, 2);

  // Final 210MB broadcast-add (bandwidth bound)
  k_fuse_out<<<BB * OUTK * NN, 128, 0, stream>>>(sHead, sTailT, Ttab, out);
}
